// BatchedCrossColumnAttentionCompressed_3427383902619
// MI455X (gfx1250) — matrix operand order repaired and checked by reading, not compiled, not run on hardware
//
#include <hip/hip_runtime.h>
#include <hip/hip_bf16.h>

typedef __attribute__((ext_vector_type(16))) __bf16 v16bf;
typedef __attribute__((ext_vector_type(8)))  float  v8f;

constexpr int Nn  = 8;
constexpr int Bb  = 4;
constexpr int Tt  = 1024;
constexpr int Dd  = 512;
constexpr int Hh  = 4;
constexpr int HDd = 128;
constexpr int Rr  = 64;
constexpr int BT  = Bb * Tt;          // 4096 rows per column n
constexpr float EPSf = 1e-5f;

__device__ __forceinline__ v8f wmma_bf16(v16bf a, v16bf b, v8f c) {
  return __builtin_amdgcn_wmma_f32_16x16x32_bf16(false, a, false, b, (short)0, c,
                                                 false, false);
}

// ---- CDNA5 async memory->LDS copies (ASYNCcnt-tracked, ISA §10.7/§15.18) ----
// INST_OFFSET is applied to both the global and the LDS address, so one
// (lds_off, gaddr) pair covers a 32B strip with offset:0 / offset:16.
__device__ __forceinline__ void async_ld_lds_b128x2(unsigned lds_off,
                                                    const void* g) {
  asm volatile("global_load_async_to_lds_b128 %0, %1, off\n\t"
               "global_load_async_to_lds_b128 %0, %1, off offset:16"
               :
               : "v"(lds_off), "v"(g)
               : "memory");
}
__device__ __forceinline__ void async_ld_lds_b128(unsigned lds_off,
                                                  const void* g) {
  asm volatile("global_load_async_to_lds_b128 %0, %1, off"
               :
               : "v"(lds_off), "v"(g)
               : "memory");
}
__device__ __forceinline__ void async_wait0() {
  asm volatile("s_wait_asynccnt 0x0" ::: "memory");
}
__device__ __forceinline__ unsigned lds_off_of(const void* p) {
  return (unsigned)(unsigned long long)p;  // flat shared addr: LDS off in [31:0]
}

// ---------------------------------------------------------------- converts
__global__ void cvt_bf16_kernel(const float* __restrict__ in,
                                __bf16* __restrict__ out, long n) {
  long i = (long)blockIdx.x * blockDim.x + threadIdx.x;
  if (i < n) out[i] = (__bf16)in[i];
}

// kW[n,r,i] = sum_o comp[n,r,o] * w[n,o,i]   (fuses projection+compression)
__global__ void fuse_comp_kernel(const float* __restrict__ comp,
                                 const float* __restrict__ w,
                                 __bf16* __restrict__ out) {
  int idx = blockIdx.x * 256 + threadIdx.x;        // over N*R*D
  int i = idx % Dd;
  int r = (idx / Dd) % Rr;
  int n = idx / (Dd * Rr);
  const float* cp = comp + ((size_t)n * Rr + r) * Dd;
  const float* wp = w + (size_t)n * Dd * Dd + i;
  float s = 0.f;
#pragma unroll 4
  for (int o = 0; o < Dd; ++o) s += cp[o] * wp[(size_t)o * Dd];
  out[idx] = (__bf16)s;
}

// ---------------------------------------------------------------- layernorm
__global__ __launch_bounds__(256)
void layernorm2_kernel(const float* __restrict__ x,
                       const float* __restrict__ wkv, const float* __restrict__ bkv,
                       const float* __restrict__ wqn, const float* __restrict__ bqn,
                       __bf16* __restrict__ okv, __bf16* __restrict__ oq) {
  int wave = threadIdx.x >> 5, lane = threadIdx.x & 31;
  int row = blockIdx.x * 8 + wave;                 // 0..N*B*T-1
  int n = row / BT;
  const float* xr = x + (size_t)row * Dd;
  float s = 0.f, s2 = 0.f;
  for (int i = lane; i < Dd; i += 32) { float v = xr[i]; s += v; s2 += v * v; }
  for (int off = 16; off > 0; off >>= 1) {
    s += __shfl_xor(s, off, 32);
    s2 += __shfl_xor(s2, off, 32);
  }
  float mean = s * (1.f / Dd);
  float var  = s2 * (1.f / Dd) - mean * mean;
  float rstd = rsqrtf(var + EPSf);
  const float* wk = wkv + (size_t)n * Dd;
  const float* bk = bkv + (size_t)n * Dd;
  const float* wq = wqn + (size_t)n * Dd;
  const float* bq = bqn + (size_t)n * Dd;
  for (int i = lane; i < Dd; i += 32) {
    float nv = (xr[i] - mean) * rstd;
    okv[(size_t)row * Dd + i] = (__bf16)(nv * wk[i] + bk[i]);
    oq [(size_t)row * Dd + i] = (__bf16)(nv * wq[i] + bq[i]);
  }
}

// --------------------------------------------------- generic WMMA GEMM
// Y[m, o] = sum_k A[m,k] * W[o,k]   (A:[M,K] row-major, W:[Nout,K] row-major)
// Block tile 128x64, K-step 32, 8 waves (4x2), wave tile 32x32 (2x2 WMMA).
enum { OUT_KC = 0, OUT_Q = 1, OUT_KV = 2, OUT_FINAL = 3 };

__global__ __launch_bounds__(256)
void gemm_wmma_kernel(const __bf16* __restrict__ A, long a_nstride,
                      const __bf16* __restrict__ W, long w_nstride,
                      int M, int K, int Nout, int mode,
                      void* __restrict__ outp,
                      const float* __restrict__ resid,
                      const unsigned char* __restrict__ mask) {
  __shared__ __align__(16) __bf16 As[128][40];
  __shared__ __align__(16) __bf16 Ws[64][40];
  __shared__ __align__(16) float  St[128][68];   // OUT_KC staging

  const int tid = threadIdx.x;
  const int lane = tid & 31, wave = tid >> 5;
  const int lh = lane >> 4, l16 = lane & 15;
  const int wm = wave >> 1, wn = wave & 1;
  const int n  = blockIdx.z;
  const int m0 = blockIdx.x * 128;
  const int n0 = blockIdx.y * 64;

  const __bf16* Abase = A + (size_t)n * a_nstride;
  const __bf16* Wbase = W + (size_t)n * w_nstride;

  v8f zero = {};
  v8f acc[2][2];
  acc[0][0] = zero; acc[0][1] = zero; acc[1][0] = zero; acc[1][1] = zero;

  const int arow = tid >> 1, acol = (tid & 1) * 16;
  const int wrow = tid >> 2, wcol = (tid & 3) * 8;
  const unsigned ldsA = lds_off_of(&As[arow][acol]);
  const unsigned ldsW = lds_off_of(&Ws[wrow][wcol]);

  for (int k0 = 0; k0 < K; k0 += 32) {
    __syncthreads();  // previous iteration's consumers done before overwrite
    const __bf16* srcA = Abase + (size_t)(m0 + arow) * K + k0 + acol;
    async_ld_lds_b128x2(ldsA, srcA);
    async_ld_lds_b128(ldsW, Wbase + (size_t)(n0 + wrow) * K + k0 + wcol);
    if (k0 + 32 < K) __builtin_prefetch(srcA + 32, 0, 1);
    async_wait0();
    __syncthreads();

    v16bf af[2], bfr[2];
#pragma unroll
    for (int i = 0; i < 2; ++i) {
      int r = wm * 32 + i * 16 + l16;
#pragma unroll
      for (int e = 0; e < 16; ++e) {
        int kk = lh * 16 + e;
        af[i][e] = As[r][kk];
      }
    }
#pragma unroll
    for (int j = 0; j < 2; ++j) {
      int c = wn * 32 + j * 16 + l16;
#pragma unroll
      for (int e = 0; e < 16; ++e) bfr[j][e] = Ws[c][lh * 16 + e];
    }
#pragma unroll
    for (int i = 0; i < 2; ++i)
#pragma unroll
      for (int j = 0; j < 2; ++j)
        acc[i][j] = wmma_bf16(af[i], bfr[j], acc[i][j]);
  }

  if (mode == OUT_KC) {
    // stage fp32 tile, then per-row int8 STE quant over R=64 (Nout==64)
#pragma unroll
    for (int i = 0; i < 2; ++i)
#pragma unroll
      for (int j = 0; j < 2; ++j)
#pragma unroll
        for (int r = 0; r < 8; ++r)
          St[wm * 32 + i * 16 + r + 8 * lh][wn * 32 + j * 16 + l16] = acc[i][j][r];
    __syncthreads();
    if (tid < 128) {
      float mf = mask[n] ? 1.f : 0.f;
      float mx = 0.f;
      for (int c = 0; c < 64; ++c) mx = fmaxf(mx, fabsf(St[tid][c] * mf));
      mx = fmaxf(mx, 1e-8f);
      float scale = 127.f / mx, inv = mx * (1.f / 127.f);
      float* op = (float*)outp + ((size_t)n * M + (m0 + tid)) * 64;
      for (int c = 0; c < 64; ++c) op[c] = rintf(St[tid][c] * mf * scale) * inv;
    }
    return;
  }

#pragma unroll
  for (int i = 0; i < 2; ++i) {
#pragma unroll
    for (int j = 0; j < 2; ++j) {
#pragma unroll
      for (int r = 0; r < 8; ++r) {
        int row = m0 + wm * 32 + i * 16 + r + 8 * lh;
        int col = n0 + wn * 32 + j * 16 + l16;
        float v = acc[i][j][r];
        if (mode == OUT_Q) {
          int b = row / Tt, t = row % Tt, h = col / HDd, hd = col % HDd;
          ((__bf16*)outp)[((((size_t)n * Bb + b) * Hh + h) * Tt + t) * HDd + hd] =
              (__bf16)v;
        } else if (mode == OUT_KV) {
          int b = row / Tt, t = row % Tt, h = col / HDd, hd = col % HDd;
          ((__bf16*)outp)[(((size_t)b * Hh + h) * Tt + t) * HDd + hd] = (__bf16)v;
        } else {  // OUT_FINAL: residual + mask select, fp32 out
          size_t idx = ((size_t)n * M + row) * (size_t)Nout + col;
          float base = resid[idx];
          ((float*)outp)[idx] = mask[n] ? base + v : base;
        }
      }
    }
  }
}

// ---------------------------------------------------------------- reduce/avg
__global__ void reduce_avg_kernel(const float* __restrict__ in,
                                  const unsigned char* __restrict__ mask,
                                  __bf16* __restrict__ out) {
  int idx = blockIdx.x * 256 + threadIdx.x;        // over BT*R
  float na = 0.f;
#pragma unroll
  for (int n = 0; n < Nn; ++n) na += mask[n] ? 1.f : 0.f;
  na = fmaxf(na, 1.f);
  float s = 0.f;
#pragma unroll
  for (int n = 0; n < Nn; ++n) s += in[(size_t)n * BT * Rr + idx];
  out[idx] = (__bf16)(s / na);
}

// ------------------------------------------------------------ flash attention
// Per (n*B+b, h): causal SDPA, Q[T,HD] vs shared K/V[T,HD], all bf16 WMMA.
// Block: 128 query rows, 8 waves x 16 rows; 32-key tiles async-staged in LDS.
__global__ __launch_bounds__(256)
void attn_kernel(const __bf16* __restrict__ qb, const __bf16* __restrict__ kb,
                 const __bf16* __restrict__ vb, __bf16* __restrict__ ob) {
  __shared__ __align__(16) __bf16 Ks[32][136];
  __shared__ __align__(16) __bf16 Vs[32][136];
  __shared__ __align__(16) __bf16 Ps[8][16][40];

  const int tid = threadIdx.x, lane = tid & 31, wave = tid >> 5;
  const int lh = lane >> 4, l16 = lane & 15;
  const int qt0 = blockIdx.x * 128;
  const int nb = blockIdx.y;            // n*B + b
  const int h  = blockIdx.z;
  const int b  = nb % Bb;

  const __bf16* qp = qb + (((size_t)nb * Hh + h) * Tt) * HDd;
  const __bf16* kp = kb + (((size_t)b  * Hh + h) * Tt) * HDd;
  const __bf16* vp = vb + (((size_t)b  * Hh + h) * Tt) * HDd;

  const int qrow = qt0 + wave * 16 + l16;
  v16bf qf[4];
#pragma unroll
  for (int ch = 0; ch < 4; ++ch)
#pragma unroll
    for (int e = 0; e < 16; ++e) {
      int kk = ch * 32 + ((e < 8) ? (lh * 8 + e) : (16 + lh * 8 + (e - 8)));
      qf[ch][e] = qp[(size_t)qrow * HDd + kk];
    }

  v8f zero = {};
  v8f oacc[8];
#pragma unroll
  for (int t8 = 0; t8 < 8; ++t8) oacc[t8] = zero;
  float mrow[8], lrow[8];
#pragma unroll
  for (int r = 0; r < 8; ++r) { mrow[r] = -1e30f; lrow[r] = 0.f; }

  const float sc = 0.088388347648318447f;  // 1/sqrt(128)
  const int krow = tid >> 3, kcolo = (tid & 7) * 16;
  const unsigned ldsK = lds_off_of(&Ks[krow][kcolo]);
  const unsigned ldsV = lds_off_of(&Vs[krow][kcolo]);
  const int nblk = (qt0 + 128) / 32;       // causal bound, uniform per block

  for (int j = 0; j < nblk; ++j) {
    const int k0 = j * 32;
    __syncthreads();
    async_ld_lds_b128x2(ldsK, kp + (size_t)(k0 + krow) * HDd + kcolo);
    async_ld_lds_b128x2(ldsV, vp + (size_t)(k0 + krow) * HDd + kcolo);
    async_wait0();
    __syncthreads();

    // S = Q K^T over HD=128 (4 k-chunks), two 16x16 tiles (32 keys)
    v8f s0 = zero, s1 = zero;
#pragma unroll
    for (int ch = 0; ch < 4; ++ch) {
      v16bf b0, b1;
#pragma unroll
      for (int e = 0; e < 16; ++e) {
        int kk = ch * 32 + ((e < 8) ? (lh * 8 + e) : (16 + lh * 8 + (e - 8)));
        b0[e] = Ks[l16][kk];
        b1[e] = Ks[16 + l16][kk];
      }
      s0 = wmma_bf16(qf[ch], b0, s0);
      s1 = wmma_bf16(qf[ch], b1, s1);
    }

    float p0[8], p1[8], cm[8], rs[8], alpha[8];
#pragma unroll
    for (int r = 0; r < 8; ++r) {
      int qr = qt0 + wave * 16 + r + 8 * lh;
      float v0 = s0[r] * sc, v1 = s1[r] * sc;
      if (k0 + l16 > qr)      v0 = -1e30f;
      if (k0 + 16 + l16 > qr) v1 = -1e30f;
      p0[r] = v0; p1[r] = v1;
      cm[r] = fmaxf(v0, v1);
    }
#pragma unroll
    for (int r = 0; r < 8; ++r)
      for (int off = 1; off < 16; off <<= 1)
        cm[r] = fmaxf(cm[r], __shfl_xor(cm[r], off, 32));
#pragma unroll
    for (int r = 0; r < 8; ++r) {
      float nm = fmaxf(mrow[r], cm[r]);
      alpha[r] = __expf(mrow[r] - nm);
      mrow[r] = nm;
      p0[r] = (p0[r] <= -1e29f) ? 0.f : __expf(p0[r] - nm);
      p1[r] = (p1[r] <= -1e29f) ? 0.f : __expf(p1[r] - nm);
      rs[r] = p0[r] + p1[r];
    }
#pragma unroll
    for (int r = 0; r < 8; ++r)
      for (int off = 1; off < 16; off <<= 1)
        rs[r] += __shfl_xor(rs[r], off, 32);
#pragma unroll
    for (int r = 0; r < 8; ++r) lrow[r] = lrow[r] * alpha[r] + rs[r];
#pragma unroll
    for (int t8 = 0; t8 < 8; ++t8)
#pragma unroll
      for (int r = 0; r < 8; ++r) oacc[t8][r] *= alpha[r];

    // C-layout P -> LDS -> A-layout fragment (per-wave staging, in-wave order)
#pragma unroll
    for (int r = 0; r < 8; ++r) {
      Ps[wave][r + 8 * lh][l16]      = (__bf16)p0[r];
      Ps[wave][r + 8 * lh][16 + l16] = (__bf16)p1[r];
    }
    v16bf pf;
#pragma unroll
    for (int e = 0; e < 16; ++e) {
      int kk = lh * 16 + e;
      pf[e] = Ps[wave][l16][kk];
    }
    // O += P V  (K=32 keys, 8 output tiles over HD)
#pragma unroll
    for (int t8 = 0; t8 < 8; ++t8) {
      v16bf bv;
#pragma unroll
      for (int e = 0; e < 16; ++e) bv[e] = Vs[lh * 16 + e][t8 * 16 + l16];
      oacc[t8] = wmma_bf16(pf, bv, oacc[t8]);
    }
  }

  float linv[8];
#pragma unroll
  for (int r = 0; r < 8; ++r) linv[r] = 1.f / lrow[r];
#pragma unroll
  for (int t8 = 0; t8 < 8; ++t8)
#pragma unroll
    for (int r = 0; r < 8; ++r) {
      int t = qt0 + wave * 16 + r + 8 * lh;
      int hd = t8 * 16 + l16;
      ob[((size_t)nb * Tt + t) * Dd + h * HDd + hd] =
          (__bf16)(oacc[t8][r] * linv[r]);
    }
}

// ---------------------------------------------------------------- launcher
extern "C" void kernel_launch(void* const* d_in, const int* in_sizes, int n_in,
                              void* d_out, int out_size, void* d_ws, size_t ws_size,
                              hipStream_t stream) {
  (void)in_sizes; (void)n_in; (void)out_size; (void)ws_size;
  const float* col    = (const float*)d_in[0];
  const unsigned char* mask = (const unsigned char*)d_in[1];
  const float* lnkw   = (const float*)d_in[2];
  const float* lnkb   = (const float*)d_in[3];
  const float* lnqw   = (const float*)d_in[4];
  const float* lnqb   = (const float*)d_in[5];
  const float* w_k    = (const float*)d_in[6];
  const float* w_v    = (const float*)d_in[7];
  const float* w_q    = (const float*)d_in[8];
  const float* w_o    = (const float*)d_in[9];
  const float* k_comp = (const float*)d_in[10];
  const float* v_comp = (const float*)d_in[11];
  const float* k_dec  = (const float*)d_in[12];
  const float* v_dec  = (const float*)d_in[13];

  char* ws = (char*)d_ws;
  size_t off = 0;
  auto alloc = [&](size_t bytes) {
    size_t o = off;
    off += (bytes + 255) & ~(size_t)255;
    return o;
  };
  const size_t NBTD = (size_t)Nn * Bb * Tt * Dd;
  __bf16* nkv   = (__bf16*)(ws + alloc(NBTD * 2));
  __bf16* nq    = (__bf16*)(ws + alloc(NBTD * 2));
  __bf16* qbuf  = (__bf16*)(ws + alloc(NBTD * 2));
  __bf16* wqb   = (__bf16*)(ws + alloc((size_t)Nn * Dd * Dd * 2));
  __bf16* wob   = (__bf16*)(ws + alloc((size_t)Nn * Dd * Dd * 2));
  __bf16* kW    = (__bf16*)(ws + alloc((size_t)Nn * Rr * Dd * 2));
  __bf16* vW    = (__bf16*)(ws + alloc((size_t)Nn * Rr * Dd * 2));
  __bf16* kdecb = (__bf16*)(ws + alloc((size_t)Dd * Rr * 2));
  __bf16* vdecb = (__bf16*)(ws + alloc((size_t)Dd * Rr * 2));
  float*  kc    = (float*) (ws + alloc((size_t)Nn * BT * Rr * 4));
  float*  vc    = (float*) (ws + alloc((size_t)Nn * BT * Rr * 4));
  __bf16* kavg  = (__bf16*)(ws + alloc((size_t)BT * Rr * 2));
  __bf16* vavg  = (__bf16*)(ws + alloc((size_t)BT * Rr * 2));
  __bf16* kbuf  = (__bf16*)(ws + alloc((size_t)Bb * Tt * Dd * 2));
  __bf16* vbuf  = (__bf16*)(ws + alloc((size_t)Bb * Tt * Dd * 2));
  __bf16* attb  = nkv;  // nkv is dead after kc/vc GEMMs; reuse for attention out

  long wcnt = (long)Nn * Dd * Dd;
  cvt_bf16_kernel<<<(int)((wcnt + 255) / 256), 256, 0, stream>>>(w_q, wqb, wcnt);
  cvt_bf16_kernel<<<(int)((wcnt + 255) / 256), 256, 0, stream>>>(w_o, wob, wcnt);
  long dcnt = (long)Dd * Rr;
  cvt_bf16_kernel<<<(int)((dcnt + 255) / 256), 256, 0, stream>>>(k_dec, kdecb, dcnt);
  cvt_bf16_kernel<<<(int)((dcnt + 255) / 256), 256, 0, stream>>>(v_dec, vdecb, dcnt);

  fuse_comp_kernel<<<(Nn * Rr * Dd) / 256, 256, 0, stream>>>(k_comp, w_k, kW);
  fuse_comp_kernel<<<(Nn * Rr * Dd) / 256, 256, 0, stream>>>(v_comp, w_v, vW);

  layernorm2_kernel<<<(Nn * BT) / 8, 256, 0, stream>>>(col, lnkw, lnkb, lnqw, lnqb,
                                                       nkv, nq);

  dim3 gKC(BT / 128, 1, Nn);
  gemm_wmma_kernel<<<gKC, 256, 0, stream>>>(nkv, (long)BT * Dd, kW, (long)Rr * Dd,
                                            BT, Dd, Rr, OUT_KC, kc, nullptr, mask);
  gemm_wmma_kernel<<<gKC, 256, 0, stream>>>(nkv, (long)BT * Dd, vW, (long)Rr * Dd,
                                            BT, Dd, Rr, OUT_KC, vc, nullptr, mask);

  reduce_avg_kernel<<<(BT * Rr) / 256, 256, 0, stream>>>(kc, mask, kavg);
  reduce_avg_kernel<<<(BT * Rr) / 256, 256, 0, stream>>>(vc, mask, vavg);

  dim3 gKV(BT / 128, Dd / 64, 1);
  gemm_wmma_kernel<<<gKV, 256, 0, stream>>>(kavg, 0, kdecb, 0, BT, Rr, Dd, OUT_KV,
                                            kbuf, nullptr, mask);
  gemm_wmma_kernel<<<gKV, 256, 0, stream>>>(vavg, 0, vdecb, 0, BT, Rr, Dd, OUT_KV,
                                            vbuf, nullptr, mask);

  dim3 gQ(BT / 128, Dd / 64, Nn);
  gemm_wmma_kernel<<<gQ, 256, 0, stream>>>(nq, (long)BT * Dd, wqb, (long)Dd * Dd,
                                           BT, Dd, Dd, OUT_Q, qbuf, nullptr, mask);

  dim3 gA(Tt / 128, Nn * Bb, Hh);
  attn_kernel<<<gA, 256, 0, stream>>>(qbuf, kbuf, vbuf, attb);

  dim3 gF(BT / 128, Dd / 64, Nn);
  gemm_wmma_kernel<<<gF, 256, 0, stream>>>(attb, (long)BT * Dd, wob, (long)Dd * Dd,
                                           BT, Dd, Dd, OUT_FINAL, d_out, col, mask);
}
